// AlbertAttention_3710851743855
// MI455X (gfx1250) — matrix operand order repaired and checked by reading, not compiled, not run on hardware
//
#include <hip/hip_runtime.h>

// ---------------------------------------------------------------------------
// ALBERT attention block on gfx1250 (MI455X), bf16 WMMA path.
// H=1024, NH=16, HD=64, B=2, S=2048  (M = B*S = 4096)
// ---------------------------------------------------------------------------

typedef __attribute__((ext_vector_type(16))) __bf16 v16bf;
typedef __attribute__((ext_vector_type(8)))  __bf16 v8bf;
typedef __attribute__((ext_vector_type(8)))  float  v8f;

union BF16x16 { v16bf v; struct { v8bf lo, hi; } p; };

#define WMMA_BF16(a, b, c) \
  __builtin_amdgcn_wmma_f32_16x16x32_bf16(false, (a), false, (b), (short)0, (c), false, false)

// CDNA5 async global->LDS copy (ASYNCcnt-tracked, LDS written by async unit)
#define ASYNC_TO_LDS_B128(ldsoff, gptr) \
  asm volatile("global_load_async_to_lds_b128 %0, %1, off" \
               :: "v"(ldsoff), "v"(gptr) : "memory")
#define WAIT_ASYNCCNT0() asm volatile("s_wait_asynccnt 0x0" ::: "memory")

// ---------------------------------------------------------------------------
// f32 -> bf16 conversion
// ---------------------------------------------------------------------------
__global__ __launch_bounds__(256)
void cvt_f32_bf16(const float* __restrict__ in, __bf16* __restrict__ out, int n) {
  int i = blockIdx.x * blockDim.x + threadIdx.x;
  if (i < n) out[i] = (__bf16)in[i];
}

// ---------------------------------------------------------------------------
// C = A[MxK] * W[NxK]^T + bias   (both operands row-major, K contiguous)
// Block = 128 threads = 4 waves. Per-wave tile 32x64 (2 m-subtiles x 4 n-tiles).
// W tile (64x32) staged once per block per k-step via async-to-LDS.
// EPI 0: f32 row-major [MxN]
// EPI 1: bf16 head layout [B,NH,S,HD]
// EPI 2: bf16 transposed head layout [B,NH,HD,S]  (for V)
// ---------------------------------------------------------------------------
template<int EPI>
__global__ __launch_bounds__(128)
void gemm_xwt(const __bf16* __restrict__ A, const __bf16* __restrict__ W,
              const float* __restrict__ bias, void* __restrict__ outp,
              int M, int N, int K, int S) {
  const int lane = threadIdx.x & 31;
  const int wave = threadIdx.x >> 5;
  const int hi   = lane >> 4;
  const int ln16 = lane & 15;

  const int m0 = blockIdx.y * 128 + wave * 32;
  const int n0 = blockIdx.x * 64;

  __shared__ __align__(16) __bf16 wtile[64 * 32];   // [n_local][k_local]

  // Cooperative staging: 2 threads per W row, 32 bytes each (2x B128)
  const int srow  = threadIdx.x >> 1;        // 0..63
  const int shalf = (threadIdx.x & 1) * 16;  // element offset 0 or 16
  const __bf16* wsrc_row = W + (size_t)(n0 + srow) * K + shalf;
  const unsigned wdst =
      (unsigned)(unsigned long long)&wtile[srow * 32 + shalf];  // LDS byte offset

  v8f acc[2][4] = {};

  for (int k = 0; k < K; k += 32) {
    // Async-stage this k-step's 64x32 W tile into LDS
    const __bf16* wsrc = wsrc_row + k;
    ASYNC_TO_LDS_B128(wdst, wsrc);
    ASYNC_TO_LDS_B128(wdst + 16u, wsrc + 8);
    __builtin_prefetch(wsrc + 32, 0, 1);     // next k-step of W -> global_prefetch_b8

    // A fragments direct from global (rows are wave-private)
    BF16x16 a[2];
#pragma unroll
    for (int mt = 0; mt < 2; ++mt) {
      const __bf16* arow = A + (size_t)(m0 + mt * 16 + ln16) * K + k + hi * 8;
      a[mt].p.lo = *(const v8bf*)(arow);
      a[mt].p.hi = *(const v8bf*)(arow + 16);
    }

    WAIT_ASYNCCNT0();
    __syncthreads();

#pragma unroll
    for (int nt = 0; nt < 4; ++nt) {
      BF16x16 b;
      const __bf16* wrow = &wtile[(nt * 16 + ln16) * 32 + hi * 8];
      b.p.lo = *(const v8bf*)(wrow);
      b.p.hi = *(const v8bf*)(wrow + 16);
#pragma unroll
      for (int mt = 0; mt < 2; ++mt)
        acc[mt][nt] = WMMA_BF16(a[mt].v, b.v, acc[mt][nt]);
    }
    __syncthreads();
  }

  // Epilogue: D layout -> VGPR r, lane l: m_local = r + 8*hi, n_local = ln16
#pragma unroll
  for (int mt = 0; mt < 2; ++mt) {
#pragma unroll
    for (int nt = 0; nt < 4; ++nt) {
      const int n = n0 + nt * 16 + ln16;
      const float bv = bias[n];
#pragma unroll
      for (int r = 0; r < 8; ++r) {
        const int m = m0 + mt * 16 + r + 8 * hi;
        const float v = acc[mt][nt][r] + bv;
        if (EPI == 0) {
          ((float*)outp)[(size_t)m * N + n] = v;
        } else if (EPI == 1) {
          const int b = m / S, s = m % S, h = n >> 6, d = n & 63;
          ((__bf16*)outp)[(((size_t)(b * 16 + h) * S) + s) * 64 + d] = (__bf16)v;
        } else {
          const int b = m / S, s = m % S, h = n >> 6, d = n & 63;
          ((__bf16*)outp)[(((size_t)(b * 16 + h) * 64) + d) * S + s] = (__bf16)v;
        }
      }
    }
  }
}

// ---------------------------------------------------------------------------
// Flash attention. Q,K in [B,NH,S,HD] bf16; V in [B,NH,HD,S] bf16 (key-major,
// so P*V B-fragments are two contiguous 16B loads -- no LDS staging for V).
// One (b,h) per blockIdx.x, 128 q-rows per blockIdx.y, 8 waves x 16 q-rows.
// ---------------------------------------------------------------------------
__global__ __launch_bounds__(256)
void attn_kernel(const __bf16* __restrict__ Qh, const __bf16* __restrict__ Kh,
                 const __bf16* __restrict__ Vt, const float* __restrict__ mask,
                 __bf16* __restrict__ Ctx) {
  const int S = 2048, HD = 64;
  const int lane = threadIdx.x & 31;
  const int wave = threadIdx.x >> 5;
  const int hi   = lane >> 4;
  const int ln16 = lane & 15;

  const int bh = blockIdx.x;      // b*16 + h
  const int b  = bh >> 4;
  const int h  = bh & 15;
  const int q0 = blockIdx.y * 128 + wave * 16;

  const __bf16* Qp = Qh + (size_t)bh * S * HD;
  const __bf16* Kp = Kh + (size_t)bh * S * HD;
  const __bf16* Vp = Vt + (size_t)bh * HD * S;   // [d][key]
  const float*  mrow = mask + (size_t)b * S;

  __shared__ __align__(16) __bf16 lds_p[8][16 * 32];

  // Q as two A fragments (d = 0..31, 32..63)
  BF16x16 aQ[2];
  {
    const __bf16* qrow = Qp + (size_t)(q0 + ln16) * HD;
#pragma unroll
    for (int t = 0; t < 2; ++t) {
      aQ[t].p.lo = *(const v8bf*)(qrow + t * 32 + hi * 8);
      aQ[t].p.hi = *(const v8bf*)(qrow + t * 32 + 16 + hi * 8);
    }
  }

  v8f O[4] = {};
  float rmax[8], rsum[8];
#pragma unroll
  for (int r = 0; r < 8; ++r) { rmax[r] = -3.0e38f; rsum[r] = 0.0f; }

  for (int j = 0; j < S; j += 32) {
    // scores = Q * K^T for two 16-key n-tiles
    v8f sc[2];
#pragma unroll
    for (int nt = 0; nt < 2; ++nt) {
      const int key = j + nt * 16 + ln16;
      const __bf16* krow = Kp + (size_t)key * HD;
      v8f z = {};
#pragma unroll
      for (int t = 0; t < 2; ++t) {
        BF16x16 bK;
        bK.p.lo = *(const v8bf*)(krow + t * 32 + hi * 8);
        bK.p.hi = *(const v8bf*)(krow + t * 32 + 16 + hi * 8);
        z = WMMA_BF16(aQ[t].v, bK.v, z);
      }
      sc[nt] = z;
    }

    // scale 1/sqrt(64) + additive mask (per key column)
#pragma unroll
    for (int nt = 0; nt < 2; ++nt) {
      const float mk = mrow[j + nt * 16 + ln16];
#pragma unroll
      for (int r = 0; r < 8; ++r) sc[nt][r] = sc[nt][r] * 0.125f + mk;
    }

    // Online softmax; rows r+8*hi reduced across each 16-lane half
    float f[8];
#pragma unroll
    for (int r = 0; r < 8; ++r) {
      float v = fmaxf(sc[0][r], sc[1][r]);
      for (int off = 8; off >= 1; off >>= 1)
        v = fmaxf(v, __shfl_xor(v, off, 16));
      const float nm = fmaxf(rmax[r], v);
      f[r] = __expf(rmax[r] - nm);
      rmax[r] = nm;
      const float p0 = __expf(sc[0][r] - nm);
      const float p1 = __expf(sc[1][r] - nm);
      sc[0][r] = p0; sc[1][r] = p1;
      float ps = p0 + p1;
      for (int off = 8; off >= 1; off >>= 1)
        ps += __shfl_xor(ps, off, 16);
      rsum[r] = rsum[r] * f[r] + ps;
#pragma unroll
      for (int dt = 0; dt < 4; ++dt) O[dt][r] *= f[r];
    }

    // Stage P (bf16, 16x32) through LDS to convert D-layout -> A-layout
#pragma unroll
    for (int nt = 0; nt < 2; ++nt)
#pragma unroll
      for (int r = 0; r < 8; ++r)
        lds_p[wave][(r + 8 * hi) * 32 + nt * 16 + ln16] = (__bf16)sc[nt][r];

    __syncthreads();  // uniform trip count; orders cross-lane LDS traffic

    BF16x16 aP;
    {
      const __bf16* pr = &lds_p[wave][ln16 * 32 + hi * 8];
      aP.p.lo = *(const v8bf*)(pr);
      aP.p.hi = *(const v8bf*)(pr + 16);
    }

    // O += P[16x32] * V[32x64]; V fragments straight from global (key-major)
#pragma unroll
    for (int dt = 0; dt < 4; ++dt) {
      BF16x16 bV;
      const __bf16* vcol = Vp + (size_t)(dt * 16 + ln16) * S + j + hi * 8;
      bV.p.lo = *(const v8bf*)(vcol);
      bV.p.hi = *(const v8bf*)(vcol + 16);
      O[dt] = WMMA_BF16(aP.v, bV.v, O[dt]);
    }

    __syncthreads();  // protect lds_p reuse next iteration
  }

  // Normalize and write ctx in [B,S,H] layout (bf16 for the final GEMM)
#pragma unroll
  for (int r = 0; r < 8; ++r) {
    const float inv = 1.0f / rsum[r];
    const int m = q0 + r + 8 * hi;
#pragma unroll
    for (int dt = 0; dt < 4; ++dt) {
      const int d = dt * 16 + ln16;
      Ctx[((size_t)(b * S + m)) * 1024 + h * 64 + d] = (__bf16)(O[dt][r] * inv);
    }
  }
}

// ---------------------------------------------------------------------------
// out = LayerNorm(hidden + proj) * gamma + beta  (block per row, H=1024)
// ---------------------------------------------------------------------------
__global__ __launch_bounds__(256)
void residual_ln(const float* __restrict__ hs, const float* __restrict__ proj,
                 const float* __restrict__ gamma, const float* __restrict__ beta,
                 float* __restrict__ out) {
  const int row = blockIdx.x;
  const float* a = hs + (size_t)row * 1024;
  const float* p = proj + (size_t)row * 1024;
  float* o = out + (size_t)row * 1024;

  __shared__ float sbuf[256], s2buf[256];
  float x[4];
  float s = 0.0f, s2 = 0.0f;
#pragma unroll
  for (int c = 0; c < 4; ++c) {
    const int i = threadIdx.x + c * 256;
    const float v = a[i] + p[i];
    x[c] = v; s += v; s2 += v * v;
  }
  sbuf[threadIdx.x] = s; s2buf[threadIdx.x] = s2;
  __syncthreads();
  for (int off = 128; off > 0; off >>= 1) {
    if (threadIdx.x < off) {
      sbuf[threadIdx.x]  += sbuf[threadIdx.x + off];
      s2buf[threadIdx.x] += s2buf[threadIdx.x + off];
    }
    __syncthreads();
  }
  const float mu  = sbuf[0] * (1.0f / 1024.0f);
  const float var = s2buf[0] * (1.0f / 1024.0f) - mu * mu;
  const float rinv = rsqrtf(var + 1e-12f);
#pragma unroll
  for (int c = 0; c < 4; ++c) {
    const int i = threadIdx.x + c * 256;
    o[i] = (x[c] - mu) * rinv * gamma[i] + beta[i];
  }
}

// ---------------------------------------------------------------------------
// Host-side orchestration
// ---------------------------------------------------------------------------
extern "C" void kernel_launch(void* const* d_in, const int* in_sizes, int n_in,
                              void* d_out, int out_size, void* d_ws, size_t ws_size,
                              hipStream_t stream) {
  const float* hs    = (const float*)d_in[0];
  const float* mask  = (const float*)d_in[1];
  const float* Wq    = (const float*)d_in[2];
  const float* bq    = (const float*)d_in[3];
  const float* Wk    = (const float*)d_in[4];
  const float* bk    = (const float*)d_in[5];
  const float* Wv    = (const float*)d_in[6];
  const float* bv    = (const float*)d_in[7];
  const float* Wd    = (const float*)d_in[8];
  const float* bd    = (const float*)d_in[9];
  const float* gamma = (const float*)d_in[10];
  const float* beta  = (const float*)d_in[11];
  float* out = (float*)d_out;

  const size_t MB = 1u << 20;
  char* ws = (char*)d_ws;
  __bf16* Xbf  = (__bf16*)(ws);             //  8 MB: X bf16 [4096 x 1024]
  __bf16* Wqb  = (__bf16*)(ws +  8 * MB);   //  2 MB
  __bf16* Wkb  = (__bf16*)(ws + 10 * MB);   //  2 MB
  __bf16* Wvb  = (__bf16*)(ws + 12 * MB);   //  2 MB
  __bf16* Wdb  = (__bf16*)(ws + 14 * MB);   //  2 MB
  __bf16* Qh   = (__bf16*)(ws + 16 * MB);   //  8 MB: [B,NH,S,HD] bf16
  __bf16* Kh   = (__bf16*)(ws + 24 * MB);   //  8 MB: [B,NH,S,HD] bf16
  __bf16* Vt   = (__bf16*)(ws + 32 * MB);   //  8 MB: [B,NH,HD,S] bf16
  __bf16* Ctx  = (__bf16*)(ws + 40 * MB);   //  8 MB: [B,S,H] bf16
  float*  Proj = (float*)(ws + 48 * MB);    // 16 MB: [B*S, H] f32

  const int M = 4096, N = 1024, K = 1024, S = 2048;

  // 1) f32 -> bf16 conversions
  cvt_f32_bf16<<<(M * K + 255) / 256, 256, 0, stream>>>(hs, Xbf, M * K);
  cvt_f32_bf16<<<(N * K + 255) / 256, 256, 0, stream>>>(Wq, Wqb, N * K);
  cvt_f32_bf16<<<(N * K + 255) / 256, 256, 0, stream>>>(Wk, Wkb, N * K);
  cvt_f32_bf16<<<(N * K + 255) / 256, 256, 0, stream>>>(Wv, Wvb, N * K);
  cvt_f32_bf16<<<(N * K + 255) / 256, 256, 0, stream>>>(Wd, Wdb, N * K);

  // 2) QKV projections (WMMA + async-LDS W staging, bias fused)
  dim3 gg(N / 64, M / 128), gb(128);
  gemm_xwt<1><<<gg, gb, 0, stream>>>(Xbf, Wqb, bq, Qh, M, N, K, S);
  gemm_xwt<1><<<gg, gb, 0, stream>>>(Xbf, Wkb, bk, Kh, M, N, K, S);
  gemm_xwt<2><<<gg, gb, 0, stream>>>(Xbf, Wvb, bv, Vt, M, N, K, S);  // V transposed

  // 3) Flash attention (32 head-batches x 16 q-chunks of 128 rows)
  attn_kernel<<<dim3(32, S / 128), 256, 0, stream>>>(Qh, Kh, Vt, mask, Ctx);

  // 4) Dense projection (WMMA, bias fused, f32 output)
  gemm_xwt<0><<<gg, gb, 0, stream>>>(Ctx, Wdb, bd, Proj, M, N, K, S);

  // 5) Residual + LayerNorm
  residual_ln<<<M, 256, 0, stream>>>(hs, Proj, gamma, beta, out);
}
